// ModulatedConv2d_42142219108369
// MI455X (gfx1250) — matrix operand order repaired and checked by reading, not compiled, not run on hardware
//
#include <hip/hip_runtime.h>
#include <math.h>

typedef __attribute__((ext_vector_type(8)))  float    v8f;
typedef __attribute__((ext_vector_type(8)))  _Float16 v8h;
typedef __attribute__((ext_vector_type(16))) _Float16 v16h;

#define NB        8
#define IN_CH     128
#define OUT_CH    128
#define HH        128
#define WW        128
#define STYLE_DIM 512
#define NTAP      9

#define TILE_W   32                // pixels per block (one row segment)
#define LDS_COLS (TILE_W + 2)      // halo of 1 on each side
#define STRH     136               // ic stride in halves: 272B -> bank step 4, 16B aligned

// lin_scale = sqrt(2/512) = 1/16 ; conv_scale = sqrt(2/(128*9)) = 1/24
#define LIN_SCALE  0.0625f
#define CONV_SCALE (1.0f / 24.0f)
#define EPS        1e-5f

// packed A halves: [B][9][c32=4][octile=8][h=2][m=16][e=16]
#define WA_ELEMS ((size_t)NB * NTAP * 4 * 8 * 2 * 16 * 16)  // 1,179,648 halves

// A-matrix 16x32 f16 register order: VGPR v (=e>>1), bit (=e&1), lane half h:
//   v<4 : K = 2v + bit + 8h ; v>=4 : K = 16 + 2(v-4) + bit + 8h
__device__ __forceinline__ int a_k_of_e(int e, int h) {
    const int v = e >> 1, bit = e & 1;
    return (v < 4) ? (2 * v + bit + 8 * h) : (16 + 2 * (v - 4) + bit + 8 * h);
}

// ---------------------------------------------------------------------------
// Kernel 1: modulation + demodulation; emit hi/lo f16 weight planes already
// swizzled into the per-lane WMMA A register order.
// ---------------------------------------------------------------------------
__global__ __launch_bounds__(128) void modulate_kernel(
    const float* __restrict__ style,   // [B, 512]
    const float* __restrict__ weight,  // [1, OC, IC, 3, 3]
    const float* __restrict__ lin_b,   // [IC]
    const float* __restrict__ lin_w,   // [IC, 512]
    _Float16* __restrict__ wAhi,
    _Float16* __restrict__ wAlo)
{
    __shared__ float s_style[STYLE_DIM];
    __shared__ float s_mod[IN_CH];

    const int b = blockIdx.x;
    const int t = threadIdx.x;

    for (int k = t; k < STYLE_DIM; k += 128)
        s_style[k] = style[b * STYLE_DIM + k];
    __syncthreads();

    {   // s[b, ic] = sum_k style[b,k] * lin_w[ic,k] * lin_scale + lin_b[ic]
        const int ic = t;
        const float* lw = lin_w + (size_t)ic * STYLE_DIM;
        float acc = lin_b[ic];
        for (int k = 0; k < STYLE_DIM; ++k)
            acc += s_style[k] * lw[k] * LIN_SCALE;
        s_mod[ic] = acc;
    }
    __syncthreads();

    const int oc = t;
    const float* wbase = weight + (size_t)oc * IN_CH * NTAP;

    float ssq = 0.0f;
    for (int ic = 0; ic < IN_CH; ++ic) {
        const float sm = s_mod[ic] * CONV_SCALE;
        #pragma unroll
        for (int tap = 0; tap < NTAP; ++tap) {
            const float wm = sm * wbase[ic * NTAP + tap];
            ssq += wm * wm;
        }
    }
    const float demod = rsqrtf(ssq + EPS);

    const int octile = oc >> 4;
    const int m      = oc & 15;
    for (int tap = 0; tap < NTAP; ++tap) {
        for (int c = 0; c < 4; ++c) {
            for (int h = 0; h < 2; ++h) {
                const size_t base =
                    (((((size_t)b * NTAP + tap) * 4 + c) * 8 + octile) * 2 + h) * 256
                    + (size_t)m * 16;
                #pragma unroll
                for (int e = 0; e < 16; ++e) {
                    const int ic = 32 * c + a_k_of_e(e, h);
                    const float wf =
                        s_mod[ic] * CONV_SCALE * demod * wbase[ic * NTAP + tap];
                    const _Float16 hi = (_Float16)wf;
                    const _Float16 lo = (_Float16)(wf - (float)hi);
                    wAhi[base + e] = hi;
                    wAlo[base + e] = lo;
                }
            }
        }
    }
}

// ---------------------------------------------------------------------------
// Kernel 2: implicit-GEMM 3x3 conv on V_WMMA_F32_16X16X32_F16 with hi/lo
// error compensation (a_hi*b_hi + a_lo*b_hi + a_hi*b_lo), fp32 accumulate.
// Block: 256 threads = 8 wave32; wave w owns oc tile [16w,16w+16).
// Block output: 128 oc x 32 pixels of one image row.
// ---------------------------------------------------------------------------
__global__ __launch_bounds__(256) void conv_wmma_kernel(
    const float* __restrict__ input,     // [B, IC, H, W]
    const _Float16* __restrict__ wAhi,   // packed A hi
    const _Float16* __restrict__ wAlo,   // packed A lo
    const float* __restrict__ bias,      // [OC]
    float* __restrict__ out)             // [B, OC, H, W]
{
    __shared__ _Float16 s_hi[3 * LDS_COLS * STRH];   // 27.7 KB
    __shared__ _Float16 s_lo[3 * LDS_COLS * STRH];   // 27.7 KB
    __shared__ float    s_bias[OUT_CH];

    const int b   = blockIdx.z;
    const int py  = blockIdx.y;
    const int px0 = blockIdx.x * TILE_W;
    const int tid = threadIdx.x;

    if (tid < OUT_CH) s_bias[tid] = bias[tid];

    // Stage input slab rows py-1..py+1, cols px0-1..px0+32, split into f16 hi/lo.
    // LDS layout: s_*[(r*LDS_COLS + c)*STRH + ic]  (ic contiguous)
    const float* inb = input + (size_t)b * IN_CH * HH * WW;
    for (int e = tid; e < 3 * IN_CH * LDS_COLS; e += 256) {
        const int r   = e / (IN_CH * LDS_COLS);
        const int rem = e - r * (IN_CH * LDS_COLS);
        const int ic  = rem / LDS_COLS;
        const int c   = rem - ic * LDS_COLS;
        const int row  = py - 1 + r;
        const int gcol = px0 - 1 + c;
        float v = 0.0f;
        if ((unsigned)row < (unsigned)HH && (unsigned)gcol < (unsigned)WW)
            v = inb[((size_t)ic * HH + row) * WW + gcol];
        const _Float16 hi = (_Float16)v;
        const _Float16 lo = (_Float16)(v - (float)hi);
        s_hi[(r * LDS_COLS + c) * STRH + ic] = hi;
        s_lo[(r * LDS_COLS + c) * STRH + ic] = lo;
    }
    __syncthreads();

    const int lane  = tid & 31;
    const int wv    = tid >> 5;    // wave id 0..7 (wave32)
    const int m     = lane & 15;   // M (A) / N (B,D) index
    const int khalf = lane >> 4;   // K half-select

    v8f acc[2] = {};

    #pragma unroll
    for (int tap = 0; tap < NTAP; ++tap) {
        const int kh = tap / 3;    // constants after unroll
        const int kw = tap - kh * 3;
        const _Float16* bh = s_hi + (kh * LDS_COLS + kw) * STRH;
        const _Float16* bl = s_lo + (kh * LDS_COLS + kw) * STRH;

        #pragma unroll
        for (int c = 0; c < 4; ++c) {   // 32-wide ic chunks
            const size_t aoff =
                (((((size_t)b * NTAP + tap) * 4 + c) * 8 + wv) * 2 + khalf) * 256
                + (size_t)m * 16;
            const v16h ahi = *(const v16h*)(wAhi + aoff);   // 32B, aligned
            const v16h alo = *(const v16h*)(wAlo + aoff);

            #pragma unroll
            for (int nt = 0; nt < 2; ++nt) {
                // B lane: N = m, K = 8*khalf .. +7, then 16 + 8*khalf .. +7
                const int boff = (16 * nt + m) * STRH + c * 32 + khalf * 8;
                const v8h h0 = *(const v8h*)(bh + boff);
                const v8h h1 = *(const v8h*)(bh + boff + 16);
                const v8h l0 = *(const v8h*)(bl + boff);
                const v8h l1 = *(const v8h*)(bl + boff + 16);
                const v16h bhi = __builtin_shufflevector(h0, h1,
                    0,1,2,3,4,5,6,7,8,9,10,11,12,13,14,15);
                const v16h blo = __builtin_shufflevector(l0, l1,
                    0,1,2,3,4,5,6,7,8,9,10,11,12,13,14,15);

                acc[nt] = __builtin_amdgcn_wmma_f32_16x16x32_f16(
                    false, ahi, false, bhi, (short)0, acc[nt], false, false);
                acc[nt] = __builtin_amdgcn_wmma_f32_16x16x32_f16(
                    false, alo, false, bhi, (short)0, acc[nt], false, false);
                acc[nt] = __builtin_amdgcn_wmma_f32_16x16x32_f16(
                    false, ahi, false, blo, (short)0, acc[nt], false, false);
            }
        }
    }

    // D layout: reg v, lane L -> M = v + 8*(L>>4), N = L&15
    #pragma unroll
    for (int nt = 0; nt < 2; ++nt) {
        const int col = px0 + 16 * nt + m;
        #pragma unroll
        for (int v = 0; v < 8; ++v) {
            const int oc = wv * 16 + v + 8 * khalf;
            out[(((size_t)b * OUT_CH + oc) * HH + py) * WW + col] =
                acc[nt][v] + s_bias[oc];
        }
    }
}

// ---------------------------------------------------------------------------
extern "C" void kernel_launch(void* const* d_in, const int* in_sizes, int n_in,
                              void* d_out, int out_size, void* d_ws, size_t ws_size,
                              hipStream_t stream) {
    const float* input  = (const float*)d_in[0]; // [8,128,128,128]
    const float* style  = (const float*)d_in[1]; // [8,512]
    const float* weight = (const float*)d_in[2]; // [1,128,128,3,3]
    const float* bias   = (const float*)d_in[3]; // [1,128,1,1]
    const float* lin_w  = (const float*)d_in[4]; // [128,512]
    const float* lin_b  = (const float*)d_in[5]; // [128]
    float* out = (float*)d_out;

    // scratch: two packed f16 weight planes (hi, lo), 2 x 2.25 MB
    _Float16* wAhi = (_Float16*)d_ws;
    _Float16* wAlo = wAhi + WA_ELEMS;

    modulate_kernel<<<dim3(NB), 128, 0, stream>>>(style, weight, lin_b, lin_w,
                                                  wAhi, wAlo);

    conv_wmma_kernel<<<dim3(WW / TILE_W, HH, NB), 256, 0, stream>>>(
        input, wAhi, wAlo, bias, out);
}
